// LinearAttention_80736795230455
// MI455X (gfx1250) — matrix operand order repaired and checked by reading, not compiled, not run on hardware
//
#include <hip/hip_runtime.h>

#define B_ 8
#define L_ 4096
#define C_ 1024
#define D_ 1024

typedef __attribute__((ext_vector_type(16))) __bf16 v16bf;
typedef __attribute__((ext_vector_type(8)))  __bf16 v8bf;
typedef __attribute__((ext_vector_type(8)))  float  v8f;
typedef __attribute__((ext_vector_type(4)))  unsigned int v4u;
typedef __attribute__((ext_vector_type(8)))  int v8i;
typedef __attribute__((ext_vector_type(4)))  int v4i;

static __device__ __forceinline__ v16bf cat16(v8bf lo, v8bf hi) {
  return __builtin_shufflevector(lo, hi, 0,1,2,3,4,5,6,7,8,9,10,11,12,13,14,15);
}

static __device__ __forceinline__ float wave_sum32(float v) {
#pragma unroll
  for (int off = 16; off > 0; off >>= 1) v += __shfl_down(v, off, 32);
  return v;
}
static __device__ __forceinline__ float wave_max32(float v) {
#pragma unroll
  for (int off = 16; off > 0; off >>= 1) v = fmaxf(v, __shfl_down(v, off, 32));
  return v;
}

// ---------------- q_probe[b][c] = mean_l Q[b][l][c] ----------------
__global__ __launch_bounds__(256) void qprobe_kernel(const float* __restrict__ Q,
                                                     float* __restrict__ qp) {
  int gid = blockIdx.x * 256 + threadIdx.x;      // B_*C_ threads
  int b = gid >> 10, c = gid & (C_ - 1);
  const float* base = Q + (size_t)b * L_ * C_ + c;
  float s = 0.f;
  for (int l = 0; l < L_; ++l) s += base[(size_t)l * C_];
  qp[gid] = s * (1.0f / (float)L_);
}

// ---------------- raw score[b][l] = q_probe[b] . K[b][l] * 1/32 ----------------
__global__ __launch_bounds__(256) void score_kernel(const float* __restrict__ Kmat,
                                                    const float* __restrict__ qp,
                                                    float* __restrict__ s) {
  int wid = blockIdx.x * 8 + (threadIdx.x >> 5);  // one wave per (b,l)
  int lane = threadIdx.x & 31;
  int b = wid >> 12;                // / L_
  int l = wid & (L_ - 1);
  const float* krow = Kmat + (size_t)b * L_ * C_ + (size_t)l * C_;
  const float* q = qp + b * C_;
  float p = 0.f;
  for (int c = lane; c < C_; c += 32) p += q[c] * krow[c];
  p = wave_sum32(p);
  if (lane == 0) s[b * L_ + l] = p * 0.03125f;   // 1/sqrt(1024)
}

// ---------------- in-place softmax over L per batch ----------------
__global__ __launch_bounds__(1024) void softmax_kernel(float* __restrict__ s) {
  __shared__ float red[32];
  __shared__ float stat[2];
  int b = blockIdx.x, tid = threadIdx.x;
  int wid = tid >> 5, lane = tid & 31;
  float* row = s + b * L_;
  float v[4];
  float mx = -3.4e38f;
#pragma unroll
  for (int i = 0; i < 4; ++i) { v[i] = row[tid + i * 1024]; mx = fmaxf(mx, v[i]); }
  mx = wave_max32(mx);
  if (lane == 0) red[wid] = mx;
  __syncthreads();
  if (tid < 32) { float m = wave_max32(red[tid]); if (tid == 0) stat[0] = m; }
  __syncthreads();
  mx = stat[0];
  float sum = 0.f;
#pragma unroll
  for (int i = 0; i < 4; ++i) { v[i] = __expf(v[i] - mx); sum += v[i]; }
  sum = wave_sum32(sum);
  __syncthreads();               // red reads (above) done before re-write
  if (lane == 0) red[wid] = sum;
  __syncthreads();
  if (tid < 32) { float t = wave_sum32(red[tid]); if (tid == 0) stat[1] = t; }
  __syncthreads();
  float inv = 1.0f / stat[1];
#pragma unroll
  for (int i = 0; i < 4; ++i) row[tid + i * 1024] = v[i] * inv;
}

// ---------------- k_sum[b][d] = sum_l phi_K[b][l][d]*w[b][l] ----------------
__global__ __launch_bounds__(256) void ksum_kernel(const float* __restrict__ phiK,
                                                   const float* __restrict__ w,
                                                   float* __restrict__ ks) {
  int gid = blockIdx.x * 256 + threadIdx.x;      // B_*D_ threads
  int b = gid >> 10, d = gid & (D_ - 1);
  const float* base = phiK + (size_t)b * L_ * D_ + d;
  const float* wb = w + b * L_;
  float s = 0.f;
  for (int l = 0; l < L_; ++l) s += base[(size_t)l * D_] * wb[l];
  ks[gid] = s;
}

// ---------------- bottom[b][l] = phi_Q[b][l] . k_sum[b] + 1e-6 ----------------
__global__ __launch_bounds__(256) void bottom_kernel(const float* __restrict__ phiQ,
                                                     const float* __restrict__ ks,
                                                     float* __restrict__ bot) {
  int wid = blockIdx.x * 8 + (threadIdx.x >> 5);
  int lane = threadIdx.x & 31;
  int b = wid >> 12;
  int l = wid & (L_ - 1);
  const float* qrow = phiQ + (size_t)b * L_ * D_ + (size_t)l * D_;
  const float* k = ks + b * D_;
  float p = 0.f;
  for (int d = lane; d < D_; d += 32) p += qrow[d] * k[d];
  p = wave_sum32(p);
  if (lane == 0) bot[b * L_ + l] = p + 1e-6f;
}

// ---------------- GEMM1: kvT[b][c][d] = sum_l (phiK[b][l][d]*w[b][l]) * V[b][l][c] ----
// 256 threads = 8 waves (2 M x 4 N); block tile 128(M=d) x 128(N=c), K-step 32
__global__ __launch_bounds__(256) void kv_gemm(const float* __restrict__ phiK,
                                               const float* __restrict__ V,
                                               const float* __restrict__ w,
                                               __bf16* __restrict__ kvT) {
  __shared__ __bf16 As[128][40];   // A[m=d][k=l] (transposed, scaled)
  __shared__ __bf16 Bs[128][40];   // Bt[n=c][k=l] (transposed)
  const int b = blockIdx.z;
  const int m0 = blockIdx.y * 128;   // D
  const int n0 = blockIdx.x * 128;   // C
  const int tid = threadIdx.x;
  const int wv = tid >> 5, lane = tid & 31;
  const int wm = wv & 1, wn = wv >> 1;
  const int g = lane >> 4, mr = lane & 15;

  const float* pk_base = phiK + (size_t)b * L_ * D_;
  const float* v_base  = V    + (size_t)b * L_ * C_;
  const float* w_base  = w + b * L_;

  v8f acc[4][2] = {};

  for (int kt = 0; kt < L_; kt += 32) {
#pragma unroll
    for (int it = 0; it < 4; ++it) {
      int idx = tid + it * 256;          // 1024 float4-chunks: 32 rows x 32 chunks
      int kk  = idx >> 5;                // k row 0..31
      int mc  = (idx & 31) << 2;         // col 0..124 step 4
      float wk = w_base[kt + kk];
      float4 a4 = *(const float4*)(pk_base + (size_t)(kt + kk) * D_ + m0 + mc);
      As[mc + 0][kk] = (__bf16)(a4.x * wk);
      As[mc + 1][kk] = (__bf16)(a4.y * wk);
      As[mc + 2][kk] = (__bf16)(a4.z * wk);
      As[mc + 3][kk] = (__bf16)(a4.w * wk);
      float4 b4 = *(const float4*)(v_base + (size_t)(kt + kk) * C_ + n0 + mc);
      Bs[mc + 0][kk] = (__bf16)b4.x;
      Bs[mc + 1][kk] = (__bf16)b4.y;
      Bs[mc + 2][kk] = (__bf16)b4.z;
      Bs[mc + 3][kk] = (__bf16)b4.w;
    }
    __syncthreads();

    v16bf afr[4], bfr[2];
#pragma unroll
    for (int i = 0; i < 4; ++i) {
      int m = wm * 64 + i * 16 + mr;
      afr[i] = cat16(*(const v8bf*)&As[m][g * 8], *(const v8bf*)&As[m][16 + g * 8]);
    }
#pragma unroll
    for (int j = 0; j < 2; ++j) {
      int n = wn * 32 + j * 16 + mr;
      bfr[j] = cat16(*(const v8bf*)&Bs[n][g * 8], *(const v8bf*)&Bs[n][16 + g * 8]);
    }
#pragma unroll
    for (int i = 0; i < 4; ++i)
#pragma unroll
      for (int j = 0; j < 2; ++j)
        acc[i][j] = __builtin_amdgcn_wmma_f32_16x16x32_bf16(
            false, afr[i], false, bfr[j], (short)0, acc[i][j], false, false);
    __syncthreads();
  }

  // write kvT[b][c][d] in bf16 (already transposed for GEMM2's B operand)
#pragma unroll
  for (int j = 0; j < 2; ++j) {
    int n = n0 + wn * 32 + j * 16 + mr;                   // c
    __bf16* dst = kvT + (size_t)b * C_ * D_ + (size_t)n * D_;
#pragma unroll
    for (int i = 0; i < 4; ++i)
#pragma unroll
      for (int r = 0; r < 8; ++r) {
        int m = m0 + wm * 64 + i * 16 + g * 8 + r;        // d
        dst[m] = (__bf16)acc[i][j][r];
      }
  }
}

// ---------------- GEMM2: out[b][l][c] = (sum_d phiQ[b][l][d]*kvT[b][c][d]) / bottom[b][l]
// B-operand tile (pure bf16 copy, row stride 1024) is fetched by the Tensor Data
// Mover straight into padded LDS (pad_interval=16 DWORDs, pad_amount=4 DWORDs ->
// 64B rows on an 80B stride), tracked with TENSORcnt.
__global__ __launch_bounds__(256) void out_gemm(const float* __restrict__ phiQ,
                                                const __bf16* __restrict__ kvT,
                                                const float* __restrict__ bot,
                                                float* __restrict__ out) {
  __shared__ __bf16 As[128][40];   // A[m=l][k=d]
  __shared__ __bf16 Bs[128][40];   // Bt[n=c][k=d]
  const int b = blockIdx.z;
  const int l0 = blockIdx.y * 128;   // L
  const int n0 = blockIdx.x * 128;   // C
  const int tid = threadIdx.x;
  const int wv = tid >> 5, lane = tid & 31;
  const int wm = wv & 1, wn = wv >> 1;
  const int g = lane >> 4, mr = lane & 15;

  const float*  a_base = phiQ + (size_t)b * L_ * D_;
  const __bf16* b_base = kvT + (size_t)b * C_ * D_;

  v8f acc[4][2] = {};

#if __has_builtin(__builtin_amdgcn_tensor_load_to_lds) && __has_builtin(__builtin_amdgcn_s_wait_tensorcnt)
  const unsigned lds_b = (unsigned)(unsigned long long)(void*)&Bs[0][0];
  // D# group1 (constant part): data_size=2B(code 1), pad_enable, pad_interval=3
  // (16 DW = 64B), pad_amount=3 (4 DW = 16B); tensor 1024x1024, tile 32x128,
  // dim0 stride 1024 elements, dim1 stride 1024*1024 (unused for 2D tile).
  const v8i g1c = { (int)((1u << 16) | (1u << 20) | (3u << 22) | (3u << 25)),
                    (int)(1024u << 16),          // tensor_dim0[15:0] in [63:48]
                    (int)(1024u << 16),          // tensor_dim0 hi=0, tensor_dim1 lo
                    (int)(32u << 16),            // tensor_dim1 hi=0, tile_dim0=32
                    (int)128,                    // tile_dim1=128, tile_dim2=0
                    (int)1024,                   // tensor_dim0_stride lo
                    (int)0,                      // stride0 hi=0, stride1 lo16=0
                    (int)16 };                   // tensor_dim1_stride hi (1<<20 >> 16)
  const v4i gz = { 0, 0, 0, 0 };
  const v8i gz8 = { 0, 0, 0, 0, 0, 0, 0, 0 };
#endif

  for (int kt = 0; kt < D_; kt += 32) {
#if __has_builtin(__builtin_amdgcn_tensor_load_to_lds) && __has_builtin(__builtin_amdgcn_s_wait_tensorcnt)
    if (wv == 0) {
      unsigned long long ga =
          (unsigned long long)(const void*)(b_base + (size_t)n0 * D_ + kt);
      v4u g0;
      g0.x = 1u;                                  // count=1, user descriptor
      g0.y = lds_b;                               // LDS byte address
      g0.z = (unsigned)ga;                        // global_addr[31:0]
      g0.w = (unsigned)(ga >> 32) | (2u << 30);   // global_addr[56:32], type=2
      __builtin_amdgcn_tensor_load_to_lds(g0, g1c, gz, gz, gz8, 0);
    }
#endif
#pragma unroll
    for (int it = 0; it < 4; ++it) {
      int idx = tid + it * 256;          // 1024 chunks: 128 rows x 8 float4
      int m   = idx >> 3;
      int kc  = (idx & 7) << 2;
      float4 a4 = *(const float4*)(a_base + (size_t)(l0 + m) * D_ + kt + kc);
      As[m][kc + 0] = (__bf16)a4.x;
      As[m][kc + 1] = (__bf16)a4.y;
      As[m][kc + 2] = (__bf16)a4.z;
      As[m][kc + 3] = (__bf16)a4.w;
    }
#if __has_builtin(__builtin_amdgcn_tensor_load_to_lds) && __has_builtin(__builtin_amdgcn_s_wait_tensorcnt)
    if (wv == 0) __builtin_amdgcn_s_wait_tensorcnt(0);
#else
#pragma unroll
    for (int it = 0; it < 2; ++it) {
      int idx = tid + it * 256;          // 512 chunks: 128 rows x 4 v8bf
      int n   = idx >> 2;
      int kc  = (idx & 3) << 3;
      *(v8bf*)&Bs[n][kc] = *(const v8bf*)(b_base + (size_t)(n0 + n) * D_ + kt + kc);
    }
#endif
    __syncthreads();

    v16bf afr[4], bfr[2];
#pragma unroll
    for (int i = 0; i < 4; ++i) {
      int m = wm * 64 + i * 16 + mr;
      afr[i] = cat16(*(const v8bf*)&As[m][g * 8], *(const v8bf*)&As[m][16 + g * 8]);
    }
#pragma unroll
    for (int j = 0; j < 2; ++j) {
      int n = wn * 32 + j * 16 + mr;
      bfr[j] = cat16(*(const v8bf*)&Bs[n][g * 8], *(const v8bf*)&Bs[n][16 + g * 8]);
    }
#pragma unroll
    for (int i = 0; i < 4; ++i)
#pragma unroll
      for (int j = 0; j < 2; ++j)
        acc[i][j] = __builtin_amdgcn_wmma_f32_16x16x32_bf16(
            false, afr[i], false, bfr[j], (short)0, acc[i][j], false, false);
    __syncthreads();  // also protects Bs against next iteration's TDM write (WAR)
  }

#pragma unroll
  for (int i = 0; i < 4; ++i)
#pragma unroll
    for (int r = 0; r < 8; ++r) {
      int l = l0 + wm * 64 + i * 16 + g * 8 + r;
      float inv = 1.0f / bot[b * L_ + l];
#pragma unroll
      for (int j = 0; j < 2; ++j) {
        int n = n0 + wn * 32 + j * 16 + mr;
        out[(size_t)b * L_ * C_ + (size_t)l * C_ + n] = acc[i][j][r] * inv;
      }
    }
}

extern "C" void kernel_launch(void* const* d_in, const int* in_sizes, int n_in,
                              void* d_out, int out_size, void* d_ws, size_t ws_size,
                              hipStream_t stream) {
  const float* Q    = (const float*)d_in[0];
  const float* Kmat = (const float*)d_in[1];
  const float* V    = (const float*)d_in[2];
  const float* phiQ = (const float*)d_in[3];
  const float* phiK = (const float*)d_in[4];
  float* out = (float*)d_out;

  float* wsf     = (float*)d_ws;
  float* w_score = wsf;                         // B_*L_
  float* q_probe = w_score + B_ * L_;           // B_*C_
  float* k_sum   = q_probe + B_ * C_;           // B_*D_
  float* bottomv = k_sum + B_ * D_;             // B_*L_
  __bf16* kvT    = (__bf16*)(bottomv + B_ * L_); // B_*C_*D_ bf16 (16 MB)

  qprobe_kernel<<<(B_ * C_) / 256, 256, 0, stream>>>(Q, q_probe);
  score_kernel<<<(B_ * L_) / 8, 256, 0, stream>>>(Kmat, q_probe, w_score);
  softmax_kernel<<<B_, 1024, 0, stream>>>(w_score);
  ksum_kernel<<<(B_ * D_) / 256, 256, 0, stream>>>(phiK, w_score, k_sum);
  bottom_kernel<<<(B_ * L_) / 8, 256, 0, stream>>>(phiQ, k_sum, bottomv);
  kv_gemm<<<dim3(C_ / 128, D_ / 128, B_), 256, 0, stream>>>(phiK, V, w_score, kvT);
  out_gemm<<<dim3(C_ / 128, L_ / 128, B_), 256, 0, stream>>>(phiQ, kvT, bottomv, out);
}
